// GATv2Stack_40699110097092
// MI455X (gfx1250) — matrix operand order repaired and checked by reading, not compiled, not run on hardware
//
#include <hip/hip_runtime.h>
#include <hip/hip_bf16.h>

// ---------------------------------------------------------------------------
// GATv2 stack forward for MI455X (gfx1250, wave32, WMMA).
// B=2, T=64, N=64, D_IN=512, HID=256, L=2, H=4, C=64  -> bt=128, M=8192
// ---------------------------------------------------------------------------

typedef __attribute__((ext_vector_type(16))) _Float16 v16h;
typedef __attribute__((ext_vector_type(8)))  float    v8f;

#define BT      128
#define NNODE   64
#define DIN     512
#define HID     256
#define NHEAD   4
#define CHEAD   64
#define MROWS   (BT * NNODE)       // 8192
#define NEG_SLOPE 0.2f
#define LN_EPS  1e-5f

// LDS row pitch (dwords) for the GEMM tiles: 20 keeps ds_load_b128 16B-aligned
// while mapping the 16 lanes of a fragment onto 16 distinct banks.
#define APITCH  20
#define BPITCH  20

__device__ __forceinline__ unsigned pack2_f16(float a, float b) {
    union { _Float16 h[2]; unsigned u; } r;
    r.h[0] = (_Float16)a;
    r.h[1] = (_Float16)b;
    return r.u;
}

// ---------------------------------------------------------------------------
// Weight pre-pack: fp32 [K][N] row-major -> u32 [K/2][N], each u32 =
// {f16(W[2k][n]), f16(W[2k+1][n])}  (WMMA B-fragment k-pair order)
// ---------------------------------------------------------------------------
__global__ void pack_weights(const float* __restrict__ W, unsigned* __restrict__ out,
                             int K, int N) {
    int idx = blockIdx.x * blockDim.x + threadIdx.x;
    int total = (K >> 1) * N;
    if (idx >= total) return;
    int kp = idx / N;
    int n  = idx - kp * N;
    float a = W[(size_t)(2 * kp) * N + n];
    float b = W[(size_t)(2 * kp + 1) * N + n];
    out[idx] = pack2_f16(a, b);
}

// ---------------------------------------------------------------------------
// GEMM: C[M][N] = A[M][K] (fp32 -> f16) * B (pre-packed f16) + bias
// block = 128 threads (4 waves), tile 128(M) x 64(N), K-step 32.
// Each wave owns a 32x64 strip: 2 A-fragments x 4 B-fragments = 8 WMMA / step.
// ---------------------------------------------------------------------------
union Frag128 { uint4 u4[2]; v16h h; };

__global__ __launch_bounds__(128)
void gemm_bias_f16(const float* __restrict__ A, const unsigned* __restrict__ Bp,
                   const float* __restrict__ bias, float* __restrict__ C,
                   int M, int N, int K) {
    __shared__ __align__(16) unsigned As[128][APITCH]; // [row][kpair] (128x32 f16)
    __shared__ __align__(16) unsigned Bs[64][BPITCH];  // [n][kpair]   (32x64 f16, transposed)

    const int t    = threadIdx.x;     // 0..127
    const int wave = t >> 5;          // 0..3
    const int lane = t & 31;
    const int bm   = blockIdx.x * 128;
    const int bn   = blockIdx.y * 64;

    v8f acc[2][4];
    #pragma unroll
    for (int mt = 0; mt < 2; ++mt)
        #pragma unroll
        for (int nt = 0; nt < 4; ++nt)
            #pragma unroll
            for (int e = 0; e < 8; ++e) acc[mt][nt][e] = 0.0f;

    const int lm  = lane & 15;        // n / m within a 16-wide subtile
    const int lhi = lane >> 4;        // 0 or 1 (K-half select)

    for (int k0 = 0; k0 < K; k0 += 32) {
        // --- stage A: 128 rows x 16 kpairs = 2048 dwords, fp32 -> packed f16 ---
        #pragma unroll
        for (int r = 0; r < 16; ++r) {
            int idx = t + r * 128;            // 0..2047
            int row = idx >> 4;
            int kp  = idx & 15;
            const float2 f = *(const float2*)(A + (size_t)(bm + row) * K + k0 + kp * 2);
            As[row][kp] = pack2_f16(f.x, f.y);
        }
        // --- stage B transposed: [n][kpair], 16 kpairs x 64 cols = 1024 dwords ---
        #pragma unroll
        for (int r = 0; r < 8; ++r) {
            int idx = t + r * 128;
            int kp  = idx >> 6;
            int n   = idx & 63;
            Bs[n][kp] = Bp[(size_t)((k0 >> 1) + kp) * N + bn + n];
        }
        // prefetch next A K-tile into cache (global_prefetch_b8)
        if (k0 + 32 < K) {
            int row = t >> 4;
            int kp  = t & 15;
            __builtin_prefetch(A + (size_t)(bm + row) * K + (k0 + 32) + kp * 2, 0, 3);
        }
        __syncthreads();

        // A fragments: lane(0-15)=rows, K 0-7 & 16-23; lane(16-31): K 8-15 & 24-31
        const int kpbase = lhi * 4;
        Frag128 af[2];
        #pragma unroll
        for (int mt = 0; mt < 2; ++mt) {
            const int mrow = (wave << 5) + (mt << 4) + lm;
            af[mt].u4[0] = *(const uint4*)&As[mrow][kpbase];
            af[mt].u4[1] = *(const uint4*)&As[mrow][kpbase + 8];
        }

        // B fragments: same K order as A (K 0-7 & 16-23 / 8-15 & 24-31) -> two ds_load_b128 each
        const int kpb = lhi * 4;
        #pragma unroll
        for (int nt = 0; nt < 4; ++nt) {
            Frag128 bf;
            const int ncol = (nt << 4) + lm;
            bf.u4[0] = *(const uint4*)&Bs[ncol][kpb];
            bf.u4[1] = *(const uint4*)&Bs[ncol][kpb + 8];
            acc[0][nt] = __builtin_amdgcn_wmma_f32_16x16x32_f16(
                false, af[0].h, false, bf.h, (short)0, acc[0][nt], false, false);
            acc[1][nt] = __builtin_amdgcn_wmma_f32_16x16x32_f16(
                false, af[1].h, false, bf.h, (short)0, acc[1][nt], false, false);
        }
        __syncthreads();
    }

    // store D (+ bias): n = lane%16, m = vgpr + (lane/16)*8
    const int mbase = lhi * 8;
    #pragma unroll
    for (int mt = 0; mt < 2; ++mt) {
        #pragma unroll
        for (int nt = 0; nt < 4; ++nt) {
            const int ncol = bn + (nt << 4) + lm;
            const float bv = bias[ncol];
            #pragma unroll
            for (int i = 0; i < 8; ++i) {
                const int m = bm + (wave << 5) + (mt << 4) + mbase + i;
                C[(size_t)m * N + ncol] = acc[mt][nt][i] + bv;
            }
        }
    }
}

// ---------------------------------------------------------------------------
// GATv2 attention layer: one block (256 threads) per graph.
// Computes g = softmax-attention(xl,xr) -> ELU -> LayerNorm -> +res -> mask
// ---------------------------------------------------------------------------
#define ATT_SMEM_FLOATS (16384 + 256*7 + 64)   // 18240 floats = 72960 bytes

__global__ __launch_bounds__(256)
void gat_attention(const float* __restrict__ xl, const float* __restrict__ xr,
                   const float* __restrict__ hin, const int* __restrict__ mask,
                   const float* __restrict__ att, const float* __restrict__ out_bias,
                   const float* __restrict__ ln_scale, const float* __restrict__ ln_bias,
                   float* __restrict__ hout) {
    extern __shared__ float smem[];
    float* xls  = smem;                 // 64*256 : xl for all nodes of this graph
    float* xri  = xls  + 64 * 256;      // 256    : xr row for current target i
    float* atts = xri  + 256;           // 256
    float* obs  = atts + 256;           // 256
    float* lnss = obs  + 256;           // 256
    float* lnbs = lnss + 256;           // 256
    float* es   = lnbs + 256;           // 256    : logits / alphas  [h*64+j]
    float* red  = es   + 256;           // 256    : reduction scratch
    int*   ms   = (int*)(red + 256);    // 64

    const int g = blockIdx.x;           // graph 0..127
    const int t = threadIdx.x;          // 0..255
    const int h = t >> 6;               // head 0..3
    const int j = t & 63;               // source node / channel

    // ---- persistent loads ----
    {
        const float4* src = (const float4*)(xl + (size_t)g * NNODE * HID);
        float4* dst = (float4*)xls;
        #pragma unroll
        for (int r = 0; r < 16; ++r) dst[t + r * 256] = src[t + r * 256];
        atts[t] = att[t];
        obs[t]  = out_bias[t];
        lnss[t] = ln_scale[t];
        lnbs[t] = ln_bias[t];
        if (t < 64) ms[t] = mask[g * 64 + t];
    }
    __syncthreads();

    const float* xrg = xr + (size_t)g * NNODE * HID;

    for (int i = 0; i < NNODE; ++i) {
        // stage xr row for target i
        xri[t] = xrg[(size_t)i * HID + t];
        __syncthreads();

        // ---- phase 1: logits e[h][j] = att_h . lrelu(xr_i + xl_j) ----
        {
            const float* xl_j = xls + (size_t)j * HID + (h << 6);
            const float* xr_i = xri + (h << 6);
            const float* a_h  = atts + (h << 6);
            float e = 0.0f;
            #pragma unroll 8
            for (int c = 0; c < CHEAD; ++c) {
                float v = xr_i[c] + xl_j[c];
                v = v > 0.0f ? v : NEG_SLOPE * v;
                e = fmaf(a_h[c], v, e);
            }
            const bool allowed = ((ms[i] != 0) && (ms[j] != 0)) || (i == j);
            e = allowed ? e : -1e9f;
            es[t]  = e;
            red[t] = e;
        }
        __syncthreads();

        // ---- phase 2: softmax over j per head (LDS tree reductions) ----
        #pragma unroll
        for (int off = 32; off > 0; off >>= 1) {
            if (j < off) red[t] = fmaxf(red[t], red[t + off]);
            __syncthreads();
        }
        const float mx = red[h << 6];
        __syncthreads();
        const float ex = __expf(es[t] - mx);
        red[t] = ex;
        __syncthreads();
        #pragma unroll
        for (int off = 32; off > 0; off >>= 1) {
            if (j < off) red[t] += red[t + off];
            __syncthreads();
        }
        const float ssum = red[h << 6];
        __syncthreads();
        es[t] = ex / ssum;              // alpha[h][j]
        __syncthreads();

        // ---- phase 3: out[i][h][c] = sum_j alpha * xl[j][h][c], t = (h,c) ----
        float o;
        {
            const int c = t & 63;
            const float* al = es + (h << 6);
            o = obs[t];
            #pragma unroll 8
            for (int j2 = 0; j2 < NNODE; ++j2)
                o = fmaf(al[j2], xls[(size_t)j2 * HID + (h << 6) + c], o);
            // ELU
            o = o > 0.0f ? o : (__expf(o) - 1.0f);
        }

        // ---- phase 4: layernorm over 256 channels of node i ----
        red[t] = o;
        __syncthreads();
        #pragma unroll
        for (int off = 128; off > 0; off >>= 1) {
            if (t < off) red[t] += red[t + off];
            __syncthreads();
        }
        const float mu = red[0] * (1.0f / 256.0f);
        __syncthreads();
        const float d = o - mu;
        red[t] = d * d;
        __syncthreads();
        #pragma unroll
        for (int off = 128; off > 0; off >>= 1) {
            if (t < off) red[t] += red[t + off];
            __syncthreads();
        }
        const float var = red[0] * (1.0f / 256.0f);
        __syncthreads();

        const float y  = d * rsqrtf(var + LN_EPS) * lnss[t] + lnbs[t];
        const size_t gi = ((size_t)g * NNODE + i) * HID + t;
        const float hv = y + hin[gi];        // residual
        hout[gi] = (ms[i] != 0) ? hv : 0.0f; // node mask
        __syncthreads();                     // es/red/xri reused next i
    }
}

// ---------------------------------------------------------------------------
// keep[g] = (#masked nodes in graph g) > 1
// ---------------------------------------------------------------------------
__global__ void compute_keep(const int* __restrict__ mask, int* __restrict__ keep) {
    int g = threadIdx.x;
    if (g < BT) {
        int s = 0;
        for (int n = 0; n < NNODE; ++n) s += (mask[g * NNODE + n] != 0);
        keep[g] = (s > 1) ? 1 : 0;
    }
}

// ---------------------------------------------------------------------------
// out = keep[g] ? h_final : h0
// ---------------------------------------------------------------------------
__global__ void final_select(const float* __restrict__ hfin, const float* __restrict__ h0,
                             const int* __restrict__ keep, float* __restrict__ out, int total) {
    int idx = blockIdx.x * blockDim.x + threadIdx.x;
    if (idx >= total) return;
    int g = idx >> 14;                 // / (64*256)
    out[idx] = keep[g] ? hfin[idx] : h0[idx];
}

// ---------------------------------------------------------------------------
// host launcher
// ---------------------------------------------------------------------------
extern "C" void kernel_launch(void* const* d_in, const int* in_sizes, int n_in,
                              void* d_out, int out_size, void* d_ws, size_t ws_size,
                              hipStream_t stream) {
    (void)in_sizes; (void)n_in; (void)out_size; (void)ws_size;

    const float* x     = (const float*)d_in[0];   // [2,64,64,512]
    const int*   pmask = (const int*)  d_in[1];   // [2,64,64] bool
    const float* W_in  = (const float*)d_in[2];   // [512,256]
    const float* b_in  = (const float*)d_in[3];   // [256]
    const float* Wl    = (const float*)d_in[4];   // [2,256,256]
    const float* bl    = (const float*)d_in[5];   // [2,256]
    const float* Wr    = (const float*)d_in[6];   // [2,256,256]
    const float* br    = (const float*)d_in[7];   // [2,256]
    const float* att   = (const float*)d_in[8];   // [2,4,64]
    const float* ob    = (const float*)d_in[9];   // [2,256]
    const float* lns   = (const float*)d_in[10];  // [2,256]
    const float* lnb   = (const float*)d_in[11];  // [2,256]

    // ---- workspace carve ----
    const size_t HM = (size_t)MROWS * HID;        // 2,097,152 floats
    float* h0 = (float*)d_ws;
    float* h1 = h0 + HM;
    float* h2 = h1 + HM;
    float* xl = h2 + HM;
    float* xr = xl + HM;
    unsigned* Wp_in = (unsigned*)(xr + HM);       // 256*256 dwords
    unsigned* Wpl0  = Wp_in + 256 * 256;          // 128*256 each
    unsigned* Wpl1  = Wpl0 + 128 * 256;
    unsigned* Wpr0  = Wpl1 + 128 * 256;
    unsigned* Wpr1  = Wpr0 + 128 * 256;
    int* keep       = (int*)(Wpr1 + 128 * 256);

    // ---- pack weights to f16 pair-dwords ----
    pack_weights<<<(256 * 256 + 255) / 256, 256, 0, stream>>>(W_in, Wp_in, DIN, HID);
    pack_weights<<<(128 * 256 + 255) / 256, 256, 0, stream>>>(Wl,             Wpl0, HID, HID);
    pack_weights<<<(128 * 256 + 255) / 256, 256, 0, stream>>>(Wl + HID * HID, Wpl1, HID, HID);
    pack_weights<<<(128 * 256 + 255) / 256, 256, 0, stream>>>(Wr,             Wpr0, HID, HID);
    pack_weights<<<(128 * 256 + 255) / 256, 256, 0, stream>>>(Wr + HID * HID, Wpr1, HID, HID);

    const dim3 gIn(MROWS / 128, HID / 64);  // (64, 4)
    const int  ATT_SMEM_BYTES = ATT_SMEM_FLOATS * (int)sizeof(float);

    // ---- input projection: h0 = x @ W_in + b_in ----
    gemm_bias_f16<<<gIn, 128, 0, stream>>>(x, Wp_in, b_in, h0, MROWS, HID, DIN);

    // ---- layer 0 ----
    gemm_bias_f16<<<gIn, 128, 0, stream>>>(h0, Wpl0, bl, xl, MROWS, HID, HID);
    gemm_bias_f16<<<gIn, 128, 0, stream>>>(h0, Wpr0, br, xr, MROWS, HID, HID);
    gat_attention<<<BT, 256, ATT_SMEM_BYTES, stream>>>(xl, xr, h0, pmask,
                                                       att, ob, lns, lnb, h1);
    // ---- layer 1 ----
    gemm_bias_f16<<<gIn, 128, 0, stream>>>(h1, Wpl1, bl + HID, xl, MROWS, HID, HID);
    gemm_bias_f16<<<gIn, 128, 0, stream>>>(h1, Wpr1, br + HID, xr, MROWS, HID, HID);
    gat_attention<<<BT, 256, ATT_SMEM_BYTES, stream>>>(xl, xr, h1, pmask,
                                                       att + NHEAD * CHEAD, ob + HID,
                                                       lns + HID, lnb + HID, h2);

    // ---- keep / fallback select ----
    compute_keep<<<1, 128, 0, stream>>>(pmask, keep);
    final_select<<<(int)((HM + 255) / 256), 256, 0, stream>>>(h2, h0, keep,
                                                              (float*)d_out, (int)HM);
}
